// CanineAttention_85950885527996
// MI455X (gfx1250) — matrix operand order repaired and checked by reading, not compiled, not run on hardware
//
#include <hip/hip_runtime.h>
#include <hip/hip_bf16.h>

// ---------------------------------------------------------------------------
// CANINE local-attention layer for gfx1250 (MI455X), bf16 WMMA pipeline.
// ---------------------------------------------------------------------------

typedef __attribute__((ext_vector_type(16))) __bf16 v16bf;
typedef __attribute__((ext_vector_type(8)))  float  v8f;

#define HID   768
#define SEQ   2048
#define BATCH 4
#define NHEAD 12
#define DH    64
#define WBLK  128
#define NBLK  16
#define MTOT  (BATCH * SEQ)       // 8192 rows

union Frag { unsigned int u[8]; v16bf v; };

__device__ __forceinline__ unsigned short f2bf(float f) {
  unsigned int u = __float_as_uint(f);
  u += 0x7fffu + ((u >> 16) & 1u);   // round-to-nearest-even
  return (unsigned short)(u >> 16);
}

// A-matrix fragment 16x32 bf16 (row-major source, ld in elements, ld even).
// lane: m = lane&15 ; kbase = (lane>>4)*8 ; V0-3 -> k=kbase+2i, V4-7 -> k=16+kbase+2i
__device__ __forceinline__ v16bf load_matA(const unsigned short* p, int ld) {
  const int lane = threadIdx.x & 31;
  const int m    = lane & 15;
  const int kb   = (lane >> 4) << 3;
  const unsigned int* row = (const unsigned int*)(p + (size_t)m * ld);
  Frag f;
#pragma unroll
  for (int i = 0; i < 4; ++i) {
    f.u[i]     = row[(kb >> 1) + i];
    f.u[i + 4] = row[((kb + 16) >> 1) + i];
  }
  return f.v;
}

// B-matrix fragment 32x16 bf16. Source: row-major [N][K] (i.e. B^T), ld even.
// lane: n = lane&15 ; kbase = (lane>>4)*8 ; V0-3 -> k=kbase+2i, V4-7 -> k=16+kbase+2i
__device__ __forceinline__ v16bf load_matB(const unsigned short* p, int ld) {
  const int lane = threadIdx.x & 31;
  const int n    = lane & 15;
  const int kb   = (lane >> 4) << 3;
  const unsigned int* row = (const unsigned int*)(p + (size_t)n * ld);
  Frag f;
#pragma unroll
  for (int i = 0; i < 4; ++i) {
    f.u[i]     = row[(kb >> 1) + i];
    f.u[i + 4] = row[((kb + 16) >> 1) + i];
  }
  return f.v;
}

__device__ __forceinline__ v8f wmma_bf16(v16bf a, v16bf b, v8f c) {
  return __builtin_amdgcn_wmma_f32_16x16x32_bf16(false, a, false, b, (short)0, c,
                                                 false, false);
}

// ---------------------------------------------------------------------------
// Conversion / transpose helpers
// ---------------------------------------------------------------------------
__global__ __launch_bounds__(256) void k_f2bf(const float* __restrict__ in,
                                              unsigned short* __restrict__ out, int n) {
  int i = blockIdx.x * 256 + threadIdx.x;
  if (i < n) out[i] = f2bf(in[i]);
}

// W [K=768][N=768] f32  ->  WT [N][K] bf16
__global__ __launch_bounds__(256) void k_transpose_w(const float* __restrict__ W,
                                                     unsigned short* __restrict__ WT) {
  int i = blockIdx.x * 256 + threadIdx.x;      // over 768*768
  int k = i / HID, n = i % HID;
  WT[(size_t)n * HID + k] = f2bf(W[(size_t)k * HID + n]);
}

// V bf16 [8192][768] -> VT bf16 [B*NH][DH][SEQ]  (token-contiguous per head-dim)
__global__ __launch_bounds__(256) void k_transpose_v(const unsigned short* __restrict__ V,
                                                     unsigned short* __restrict__ VT) {
  int i = blockIdx.x * 256 + threadIdx.x;      // over 8192*768
  int t = i / HID, c = i % HID;
  int b = t >> 11, tt = t & (SEQ - 1);
  int h = c >> 6,  d  = c & (DH - 1);
  VT[(((size_t)(b * NHEAD + h) * DH + d) << 11) + tt] = V[i];
}

// ---------------------------------------------------------------------------
// GEMM: C[8192 x 768] = A(bf16) x B(bf16, given as B^T [N][K]) + bias
// 256 threads = 8 waves; wave grid 4(rows)x2(cols); wave tile 32x64.
// If outBf: store bf16 (acc+bias)*scale.  Else: store f32 acc+bias+residual.
// Next K-step tiles prefetched (WGP-scope global_prefetch_b8) while WMMAs drain.
// ---------------------------------------------------------------------------
__global__ __launch_bounds__(256) void k_gemm(const unsigned short* __restrict__ A,
                                              const unsigned short* __restrict__ BT,
                                              const float* __restrict__ bias,
                                              const float* __restrict__ residual,
                                              float* __restrict__ outF,
                                              unsigned short* __restrict__ outBf,
                                              float scale) {
  const int wv   = threadIdx.x >> 5;
  const int lane = threadIdx.x & 31;
  const int lh   = lane >> 4;
  const int ln   = lane & 15;
  const int rowbase = blockIdx.y * 128 + (wv & 3) * 32;
  const int colbase = blockIdx.x * 128 + (wv >> 2) * 64;

  v8f acc[2][4] = {};
  for (int kk = 0; kk < HID; kk += 32) {
    const unsigned short* a0p = A + (size_t)rowbase * HID + kk;
    const unsigned short* a1p = A + (size_t)(rowbase + 16) * HID + kk;
    if (kk + 32 < HID) {
      // pull next K-step into the near (WGP) cache while this step computes
      __builtin_prefetch(a0p + 32 + (size_t)(lane & 15) * HID, 0, 3);
      __builtin_prefetch(BT + (size_t)(colbase + (lane & 15)) * HID + kk + 32, 0, 3);
    }
    v16bf a0 = load_matA(a0p, HID);
    v16bf a1 = load_matA(a1p, HID);
#pragma unroll
    for (int nt = 0; nt < 4; ++nt) {
      v16bf bb = load_matB(BT + (size_t)(colbase + nt * 16) * HID + kk, HID);
      acc[0][nt] = wmma_bf16(a0, bb, acc[0][nt]);
      acc[1][nt] = wmma_bf16(a1, bb, acc[1][nt]);
    }
  }

#pragma unroll
  for (int mt = 0; mt < 2; ++mt)
#pragma unroll
    for (int nt = 0; nt < 4; ++nt)
#pragma unroll
      for (int v = 0; v < 8; ++v) {
        int m = rowbase + mt * 16 + v + 8 * lh;
        int n = colbase + nt * 16 + ln;
        float val = acc[mt][nt][v] + bias[n];
        if (outBf) {
          outBf[(size_t)m * HID + n] = f2bf(val * scale);
        } else {
          outF[(size_t)m * HID + n] = val + residual[(size_t)m * HID + n];
        }
      }
}

// ---------------------------------------------------------------------------
// Block-diagonal attention: one workgroup per (block, head, batch).
// 4 waves, each owns 32 query rows: scores(128xK64) -> softmax -> P*V(K128).
// ---------------------------------------------------------------------------
__global__ __launch_bounds__(128) void k_attn(const unsigned short* __restrict__ Qbf,
                                              const unsigned short* __restrict__ Kbf,
                                              const unsigned short* __restrict__ VT,
                                              const float* __restrict__ mask,
                                              unsigned short* __restrict__ Ctx) {
  const int blk = blockIdx.x;        // 0..15
  const int h   = blockIdx.y;        // 0..11
  const int b   = blockIdx.z;        // 0..3
  const int wv  = threadIdx.x >> 5;  // 0..3
  const int lane = threadIdx.x & 31;
  const int lh = lane >> 4;
  const int ln = lane & 15;

  const int t0      = b * SEQ + blk * WBLK;   // global token row base
  const int rowbase = wv * 32;

  __shared__ __align__(16) unsigned short sP[128 * 136];

  const unsigned short* vbase =
      VT + (((size_t)(b * NHEAD + h) * DH) << 11) + blk * WBLK;
  // pull the V tile (used after softmax) into the WGP cache while QK^T runs
  __builtin_prefetch(vbase + ((size_t)(threadIdx.x & 63) << 11), 0, 3);

  // ---- scores = Qs (pre-scaled by 1/8) x K^T -------------------------------
  v16bf aq[2][2];
#pragma unroll
  for (int mt = 0; mt < 2; ++mt)
#pragma unroll
    for (int ks = 0; ks < 2; ++ks)
      aq[mt][ks] = load_matA(
          Qbf + (size_t)(t0 + rowbase + mt * 16) * HID + h * DH + ks * 32, HID);

  v8f sc[2][8] = {};
#pragma unroll
  for (int nt = 0; nt < 8; ++nt)
#pragma unroll
    for (int ks = 0; ks < 2; ++ks) {
      v16bf bk = load_matB(
          Kbf + (size_t)(t0 + nt * 16) * HID + h * DH + ks * 32, HID);
      sc[0][nt] = wmma_bf16(aq[0][ks], bk, sc[0][nt]);
      sc[1][nt] = wmma_bf16(aq[1][ks], bk, sc[1][nt]);
    }

  // ---- mask bias + row softmax (rows live in 16-lane halves) ---------------
  const float* mb = mask + ((size_t)b << 22);   // b * S * S
  float rsum[2][8];
#pragma unroll
  for (int mt = 0; mt < 2; ++mt)
#pragma unroll
    for (int v = 0; v < 8; ++v) {
      int q = blk * WBLK + rowbase + mt * 16 + v + 8 * lh;   // within batch
      float mx = -1e30f;
#pragma unroll
      for (int nt = 0; nt < 8; ++nt) {
        int kc = blk * WBLK + nt * 16 + ln;
        float s = sc[mt][nt][v] + (1.0f - mb[((size_t)q << 11) + kc]) * -10000.0f;
        sc[mt][nt][v] = s;
        mx = fmaxf(mx, s);
      }
#pragma unroll
      for (int d = 1; d < 16; d <<= 1) mx = fmaxf(mx, __shfl_xor(mx, d, 32));
      float sum = 0.0f;
#pragma unroll
      for (int nt = 0; nt < 8; ++nt) {
        float p = __expf(sc[mt][nt][v] - mx);
        sc[mt][nt][v] = p;
        sum += p;
      }
#pragma unroll
      for (int d = 1; d < 16; d <<= 1) sum += __shfl_xor(sum, d, 32);
      rsum[mt][v] = sum;
      int mrow = rowbase + mt * 16 + v + 8 * lh;
#pragma unroll
      for (int nt = 0; nt < 8; ++nt)
        sP[mrow * 136 + nt * 16 + ln] = f2bf(sc[mt][nt][v]);
    }
  __syncthreads();

  // ---- ctx = P x V ---------------------------------------------------------
  v8f ct[2][4] = {};
#pragma unroll
  for (int ks = 0; ks < 4; ++ks) {
    v16bf ap0 = load_matA(sP + (rowbase)*136 + ks * 32, 136);
    v16bf ap1 = load_matA(sP + (rowbase + 16) * 136 + ks * 32, 136);
#pragma unroll
    for (int nt = 0; nt < 4; ++nt) {
      v16bf bv = load_matB(vbase + ((size_t)(nt * 16) << 11) + ks * 32, SEQ);
      ct[0][nt] = wmma_bf16(ap0, bv, ct[0][nt]);
      ct[1][nt] = wmma_bf16(ap1, bv, ct[1][nt]);
    }
  }

#pragma unroll
  for (int mt = 0; mt < 2; ++mt)
#pragma unroll
    for (int nt = 0; nt < 4; ++nt)
#pragma unroll
      for (int v = 0; v < 8; ++v) {
        int trow = t0 + rowbase + mt * 16 + v + 8 * lh;
        float val = ct[mt][nt][v] / rsum[mt][v];
        Ctx[(size_t)trow * HID + h * DH + nt * 16 + ln] = f2bf(val);
      }
}

// ---------------------------------------------------------------------------
// Row LayerNorm over 768 (eps = 1e-12), one block per row.
// ---------------------------------------------------------------------------
__global__ __launch_bounds__(256) void k_ln(const float* __restrict__ y,
                                            const float* __restrict__ gam,
                                            const float* __restrict__ bet,
                                            float* __restrict__ out) {
  const int row = blockIdx.x;
  const int tid = threadIdx.x;
  const float* x = y + (size_t)row * HID;
  float v0 = x[tid], v1 = x[tid + 256], v2 = x[tid + 512];
  __shared__ float ss[256], sq[256];
  ss[tid] = v0 + v1 + v2;
  sq[tid] = v0 * v0 + v1 * v1 + v2 * v2;
  __syncthreads();
  for (int o = 128; o > 0; o >>= 1) {
    if (tid < o) { ss[tid] += ss[tid + o]; sq[tid] += sq[tid + o]; }
    __syncthreads();
  }
  float mu  = ss[0] * (1.0f / HID);
  float var = sq[0] * (1.0f / HID) - mu * mu;
  float inv = rsqrtf(var + 1e-12f);
  float* o0 = out + (size_t)row * HID;
  o0[tid]       = (v0 - mu) * inv * gam[tid]       + bet[tid];
  o0[tid + 256] = (v1 - mu) * inv * gam[tid + 256] + bet[tid + 256];
  o0[tid + 512] = (v2 - mu) * inv * gam[tid + 512] + bet[tid + 512];
}

// ---------------------------------------------------------------------------
extern "C" void kernel_launch(void* const* d_in, const int* in_sizes, int n_in,
                              void* d_out, int out_size, void* d_ws, size_t ws_size,
                              hipStream_t stream) {
  const float* X  = (const float*)d_in[0];
  const float* AM = (const float*)d_in[1];
  const float* Wq = (const float*)d_in[2];  const float* bq = (const float*)d_in[3];
  const float* Wk = (const float*)d_in[4];  const float* bk = (const float*)d_in[5];
  const float* Wv = (const float*)d_in[6];  const float* bv = (const float*)d_in[7];
  const float* Wo = (const float*)d_in[8];  const float* bo = (const float*)d_in[9];
  const float* g  = (const float*)d_in[10]; const float* be = (const float*)d_in[11];
  float* out = (float*)d_out;

  const size_t NX = (size_t)MTOT * HID;   // 6,291,456
  const size_t NW = (size_t)HID * HID;    //   589,824

  unsigned short* ws16 = (unsigned short*)d_ws;
  unsigned short* Xbf = ws16;                 // NX
  unsigned short* WqT = Xbf + NX;             // NW
  unsigned short* WkT = WqT + NW;
  unsigned short* WvT = WkT + NW;
  unsigned short* WoT = WvT + NW;
  unsigned short* Qbf = WoT + NW;             // NX each
  unsigned short* Kbf = Qbf + NX;
  unsigned short* Vbf = Kbf + NX;
  unsigned short* VT  = Vbf + NX;
  unsigned short* Ctx = VT + NX;
  float*          Y   = (float*)(Ctx + NX);   // NX f32

  k_f2bf<<<(int)(NX / 256), 256, 0, stream>>>(X, Xbf, (int)NX);
  k_transpose_w<<<(int)(NW / 256), 256, 0, stream>>>(Wq, WqT);
  k_transpose_w<<<(int)(NW / 256), 256, 0, stream>>>(Wk, WkT);
  k_transpose_w<<<(int)(NW / 256), 256, 0, stream>>>(Wv, WvT);
  k_transpose_w<<<(int)(NW / 256), 256, 0, stream>>>(Wo, WoT);

  dim3 gg(HID / 128, MTOT / 128);             // (6, 64)
  k_gemm<<<gg, 256, 0, stream>>>(Xbf, WqT, bq, nullptr, nullptr, Qbf, 0.125f);
  k_gemm<<<gg, 256, 0, stream>>>(Xbf, WkT, bk, nullptr, nullptr, Kbf, 1.0f);
  k_gemm<<<gg, 256, 0, stream>>>(Xbf, WvT, bv, nullptr, nullptr, Vbf, 1.0f);

  k_transpose_v<<<(int)(NX / 256), 256, 0, stream>>>(Vbf, VT);

  k_attn<<<dim3(NBLK, NHEAD, BATCH), 128, 0, stream>>>(Qbf, Kbf, VT, AM, Ctx);

  k_gemm<<<gg, 256, 0, stream>>>(Ctx, WoT, bo, X, Y, nullptr, 1.0f);

  k_ln<<<MTOT, 256, 0, stream>>>(Y, g, be, out);
}
